// GCN_53523882442951
// MI455X (gfx1250) — matrix operand order repaired and checked by reading, not compiled, not run on hardware
//
#include <hip/hip_runtime.h>
#include <hip/hip_bf16.h>

#define N_NODES  100000
#define N_EDGES  1600000
#define N_GRAPHS 2048

typedef __attribute__((ext_vector_type(16))) __bf16 v16bf;
typedef __attribute__((ext_vector_type(8)))  float  v8f;

// ---------------- degree / norm ----------------
__global__ void k_init_deg(float* __restrict__ deg) {
  int i = blockIdx.x * blockDim.x + threadIdx.x;
  if (i < N_NODES) deg[i] = 1.0f;
}
__global__ void k_accum_deg(const int* __restrict__ dst, float* __restrict__ deg) {
  int e = blockIdx.x * blockDim.x + threadIdx.x;
  if (e < N_EDGES) atomicAdd(&deg[dst[e]], 1.0f);
}
__global__ void k_rsqrt_deg(const float* __restrict__ deg, float* __restrict__ dinv) {
  int i = blockIdx.x * blockDim.x + threadIdx.x;
  if (i < N_NODES) dinv[i] = rsqrtf(deg[i]);
}

// ---------------- WMMA GEMM: Y[R x N] = bf16(X[R x K]) @ bf16(W[K x N]) ----------------
// One wave per 16x16 output tile; K,N compile-time so the K-loop fully unrolls
// into back-to-back v_wmma ops with float4 (b128) fragment loads.
template <int K, int N>
__global__ void k_gemm_wmma_bf16(const float* __restrict__ X, const float* __restrict__ W,
                                 float* __restrict__ Y, int ntiles) {
  int wt = blockIdx.x * 8 + (threadIdx.x >> 5);          // 8 waves / 256-thread block
  if (wt >= ntiles) return;                               // wave-uniform guard
  constexpr int CT = N >> 4;
  int rt = wt / CT;
  int ct = wt - rt * CT;

  int lane = threadIdx.x & 31;
  int half = lane >> 4;          // 0: lanes 0-15, 1: lanes 16-31
  int l15  = lane & 15;

  int arow = rt * 16 + l15;      // A-fragment row for this lane
  int col0 = ct * 16;

  v8f c = {};
#pragma unroll
  for (int kk = 0; kk < K; kk += 32) {
    v16bf a, b;
    // A 16x32 bf16 fragment (ISA 7.12.2): lane half selects K offset 8*half;
    // elements 0..7 -> K = kk+8*half+0..7, elements 8..15 -> K = kk+8*half+16..23
    const float4* xr = (const float4*)(X + (size_t)arow * K + kk + 8 * half);
    float4 a0 = xr[0], a1 = xr[1], a2 = xr[4], a3 = xr[5];
    a[0]  = (__bf16)a0.x; a[1]  = (__bf16)a0.y; a[2]  = (__bf16)a0.z; a[3]  = (__bf16)a0.w;
    a[4]  = (__bf16)a1.x; a[5]  = (__bf16)a1.y; a[6]  = (__bf16)a1.z; a[7]  = (__bf16)a1.w;
    a[8]  = (__bf16)a2.x; a[9]  = (__bf16)a2.y; a[10] = (__bf16)a2.z; a[11] = (__bf16)a2.w;
    a[12] = (__bf16)a3.x; a[13] = (__bf16)a3.y; a[14] = (__bf16)a3.z; a[15] = (__bf16)a3.w;

    // B 32x16 bf16 fragment: lane holds col n = col0 + l15; elements 0..7 -> K = kk+8*half+0..7,
    // elements 8..15 -> K = kk+8*half+16..23
    const float* wp = W + (size_t)(kk + 8 * half) * N + col0 + l15;
#pragma unroll
    for (int s = 0; s < 8; ++s) {
      b[s]     = (__bf16)wp[(size_t)s * N];
      b[s + 8] = (__bf16)wp[(size_t)(s + 16) * N];
    }

    c = __builtin_amdgcn_wmma_f32_16x16x32_bf16(false, a, false, b, (short)0, c,
                                                false, false);
  }
  // C/D layout: VGPR v -> row (v + 8*half), lane l15 -> col
  float* yr = Y + (size_t)(rt * 16 + 8 * half) * N + col0 + l15;
#pragma unroll
  for (int v = 0; v < 8; ++v) yr[(size_t)v * N] = c[v];
}

// ---------------- self-loop init: A = G * dinv^2 ----------------
__global__ void k_self_init(const float* __restrict__ G, const float* __restrict__ dinv,
                            float* __restrict__ A, int fshift, int total) {
  int i = blockIdx.x * blockDim.x + threadIdx.x;
  if (i >= total) return;
  int node = i >> fshift;
  float d = dinv[node];
  A[i] = G[i] * d * d;
}

// ---------------- edge scatter: A[dst] += G[src] * dinv[src]*dinv[dst] ----------------
// one thread per (edge, 4-feature group); F/4 = 1<<fq
__global__ void k_edge_scatter(const int* __restrict__ src, const int* __restrict__ dst,
                               const float* __restrict__ dinv, const float* __restrict__ G,
                               float* __restrict__ A, int F, int fq, long total) {
  long i = (long)blockIdx.x * blockDim.x + threadIdx.x;
  if (i >= total) return;
  int e  = (int)(i >> fq);
  int f4 = (int)(i & ((1 << fq) - 1));
  int s = src[e], d = dst[e];
  float coef = dinv[s] * dinv[d];
  float4 v = ((const float4*)(G + (size_t)s * F))[f4];
  float* ap = A + (size_t)d * F + 4 * f4;
  atomicAdd(ap + 0, v.x * coef);
  atomicAdd(ap + 1, v.y * coef);
  atomicAdd(ap + 2, v.z * coef);
  atomicAdd(ap + 3, v.w * coef);
}

// ---------------- bias + optional relu (in place) ----------------
__global__ void k_bias_act(float* __restrict__ A, const float* __restrict__ b,
                           int F, int do_relu, int total) {
  int i = blockIdx.x * blockDim.x + threadIdx.x;
  if (i >= total) return;
  float v = A[i] + b[i & (F - 1)];
  A[i] = do_relu ? fmaxf(v, 0.0f) : v;
}

// ---------------- mean pool ----------------
__global__ void k_pool_zero(float* __restrict__ pooled, float* __restrict__ cnt) {
  int i = blockIdx.x * blockDim.x + threadIdx.x;
  if (i < N_GRAPHS * 32) pooled[i] = 0.0f;
  if (i < N_GRAPHS) cnt[i] = 0.0f;
}
__global__ void k_pool_accum(const float* __restrict__ H, const int* __restrict__ batch,
                             float* __restrict__ pooled, float* __restrict__ cnt) {
  int i = blockIdx.x * blockDim.x + threadIdx.x;
  if (i >= N_NODES * 32) return;
  int node = i >> 5, f = i & 31;
  int g = batch[node];
  atomicAdd(&pooled[g * 32 + f], H[i]);
  if (f == 0) atomicAdd(&cnt[g], 1.0f);
}

// ---------------- head: logits, sigmoid, BCE loss ----------------
__global__ void k_head(const float* __restrict__ pooled, const float* __restrict__ cnt,
                       const float* __restrict__ Wl, const float* __restrict__ bl,
                       const int* __restrict__ targets, float* __restrict__ out) {
  __shared__ float red[256];
  float lsum = 0.0f;
  for (int g = threadIdx.x; g < N_GRAPHS; g += 256) {
    float c = fmaxf(cnt[g], 1.0f);
    float acc = 0.0f;
#pragma unroll
    for (int f = 0; f < 32; ++f) acc += (pooled[g * 32 + f] / c) * Wl[f];
    float logit = acc + bl[0];
    out[g] = 1.0f / (1.0f + expf(-logit));
    float y = (float)targets[g];
    lsum += fmaxf(logit, 0.0f) - logit * y + log1pf(expf(-fabsf(logit)));
  }
  red[threadIdx.x] = lsum;
  __syncthreads();
  for (int s = 128; s > 0; s >>= 1) {
    if (threadIdx.x < s) red[threadIdx.x] += red[threadIdx.x + s];
    __syncthreads();
  }
  if (threadIdx.x == 0) out[N_GRAPHS] = red[0] / (float)N_GRAPHS;
}

// ---------------- host side ----------------
static inline int cdiv(long a, int b) { return (int)((a + b - 1) / b); }

extern "C" void kernel_launch(void* const* d_in, const int* in_sizes, int n_in,
                              void* d_out, int out_size, void* d_ws, size_t ws_size,
                              hipStream_t stream) {
  const float* x       = (const float*)d_in[0];
  const int*   eidx    = (const int*)d_in[1];
  const int*   batch   = (const int*)d_in[2];
  const int*   targets = (const int*)d_in[3];
  const float* W1 = (const float*)d_in[4];
  const float* b1 = (const float*)d_in[5];
  const float* W2 = (const float*)d_in[6];
  const float* b2 = (const float*)d_in[7];
  const float* W3 = (const float*)d_in[8];
  const float* b3 = (const float*)d_in[9];
  const float* Wl = (const float*)d_in[10];
  const float* bl = (const float*)d_in[11];
  float* out = (float*)d_out;

  const int* src = eidx;            // edge_index[0]
  const int* dst = eidx + N_EDGES;  // edge_index[1]

  // workspace carve-up (all offsets 512B aligned)
  char* ws = (char*)d_ws;
  float* deg    = (float*)(ws + 0);                 // 100000 f
  float* dinv   = (float*)(ws + 401408);            // 100000 f
  float* pooled = (float*)(ws + 802816);            // 2048*32 f
  float* cnt    = (float*)(ws + 1064960);           // 2048 f
  float* bufA   = (float*)(ws + 1073152);           // 100000*128 f
  float* bufB   = (float*)(ws + 1073152 + 51200000);// 100000*128 f

  const int T = 256;

  // degrees / symmetric norm
  k_init_deg<<<cdiv(N_NODES, T), T, 0, stream>>>(deg);
  k_accum_deg<<<cdiv(N_EDGES, T), T, 0, stream>>>(dst, deg);
  k_rsqrt_deg<<<cdiv(N_NODES, T), T, 0, stream>>>(deg, dinv);

  const int RT = N_NODES / 16; // 6250 row tiles

  // ---- layer 1: K=128, N=128, relu ----
  {
    constexpr int F = 128;
    int tiles = RT * (F / 16);
    k_gemm_wmma_bf16<128, 128><<<cdiv(tiles, 8), T, 0, stream>>>(x, W1, bufA, tiles);
    k_self_init<<<cdiv((long)N_NODES * F, T), T, 0, stream>>>(bufA, dinv, bufB, 7,
                                                              N_NODES * F);
    long tot = (long)N_EDGES * (F / 4);
    k_edge_scatter<<<cdiv(tot, T), T, 0, stream>>>(src, dst, dinv, bufA, bufB, F, 5, tot);
    k_bias_act<<<cdiv((long)N_NODES * F, T), T, 0, stream>>>(bufB, b1, F, 1, N_NODES * F);
  }
  // ---- layer 2: K=128, N=64, relu ----  (in: bufB stride 128, out: bufA stride 64)
  {
    constexpr int F = 64;
    int tiles = RT * (F / 16);
    k_gemm_wmma_bf16<128, 64><<<cdiv(tiles, 8), T, 0, stream>>>(bufB, W2, bufA, tiles);
    k_self_init<<<cdiv((long)N_NODES * F, T), T, 0, stream>>>(bufA, dinv, bufB, 6,
                                                              N_NODES * F);
    long tot = (long)N_EDGES * (F / 4);
    k_edge_scatter<<<cdiv(tot, T), T, 0, stream>>>(src, dst, dinv, bufA, bufB, F, 4, tot);
    k_bias_act<<<cdiv((long)N_NODES * F, T), T, 0, stream>>>(bufB, b2, F, 1, N_NODES * F);
  }
  // ---- layer 3: K=64, N=32, no relu ----
  {
    constexpr int F = 32;
    int tiles = RT * (F / 16);
    k_gemm_wmma_bf16<64, 32><<<cdiv(tiles, 8), T, 0, stream>>>(bufB, W3, bufA, tiles);
    k_self_init<<<cdiv((long)N_NODES * F, T), T, 0, stream>>>(bufA, dinv, bufB, 5,
                                                              N_NODES * F);
    long tot = (long)N_EDGES * (F / 4);
    k_edge_scatter<<<cdiv(tot, T), T, 0, stream>>>(src, dst, dinv, bufA, bufB, F, 3, tot);
    k_bias_act<<<cdiv((long)N_NODES * F, T), T, 0, stream>>>(bufB, b3, F, 0, N_NODES * F);
  }

  // ---- mean pool + head ----
  k_pool_zero<<<cdiv(N_GRAPHS * 32, T), T, 0, stream>>>(pooled, cnt);
  k_pool_accum<<<cdiv((long)N_NODES * 32, T), T, 0, stream>>>(bufB, batch, pooled, cnt);
  k_head<<<1, T, 0, stream>>>(pooled, cnt, Wl, bl, targets, out);
}
